// MotionDecoder_36309653520632
// MI455X (gfx1250) — matrix operand order repaired and checked by reading, not compiled, not run on hardware
//
#include <hip/hip_runtime.h>
#include <cmath>

// ---------------------------------------------------------------------------
// MotionDecoder fused kernel for MI455X (gfx1250, wave32, WMMA).
//   h   = GELU(x[b,t,j,:] @ W1[j] + b1[j])        (D=512 -> D=512)
//   out = h @ W2[j] + b2[j]                       (D=512 -> 6 or 3), concat
// Strategy: bf16 V_WMMA_F32_16X16X32_BF16 with fp32 accumulation.
//   pass 1: swizzle W1/W2 fp32 -> bf16 in exact WMMA B-fragment order (d_ws)
//   pass 2: per (joint, 32-row tile): stage X tile in LDS (bf16), layer-1
//           WMMA, GELU, h->LDS, layer-2 WMMA (N padded to 16), reduce, store.
// ---------------------------------------------------------------------------

typedef __attribute__((ext_vector_type(16))) __bf16        v16bf;
typedef __attribute__((ext_vector_type(8)))  float         v8f;
typedef __attribute__((ext_vector_type(4)))  float         f32x4;
typedef __attribute__((ext_vector_type(4)))  unsigned int  u32x4;
typedef __attribute__((ext_vector_type(2)))  unsigned int  u32x2;

#define J6_CNT   13
#define J_TOT    43
#define D_DIM    512
#define ROWS     3136          // B*T = 16*196
#define MTILE    32
#define NROWT    98            // 3136 / 32, exact
#define OUT_STR  168           // 13*6 + 30*3
#define RSTRIDE  1040          // LDS row stride bytes: (512+8) bf16, conflict-free

__device__ __forceinline__ unsigned int f2bf(float f) {
  unsigned int u = __float_as_uint(f);
  return (u + 0x7FFFu + ((u >> 16) & 1u)) >> 16;          // round-to-nearest-even
}
__device__ __forceinline__ unsigned int pack_bf(float lo, float hi) {
  return f2bf(lo) | (f2bf(hi) << 16);
}

union Frag { v16bf v; u32x4 q[2]; };

// --------------------------------------------------------------------------
// W1 fp32[J,512,512] -> ws1 bf16 packed in B-fragment order.
// B 32x16 16-bit layout (contiguous k, not ISA table order): lanes 0-15: n=lane, V_v = K{2v,2v+1};
// lanes 16-31: n=lane-16, V_v = K{16+2v,17+2v}.
// dword index = (((j*32 + ntile)*16 + kstep)*256 + lane*8 + v)
// --------------------------------------------------------------------------
__global__ void swizzle_w1(const float* __restrict__ W1, unsigned int* __restrict__ ws1) {
  unsigned int idx  = blockIdx.x * 256u + threadIdx.x;     // 43*32*16*256 total
  unsigned int v    = idx & 7u;
  unsigned int lane = (idx >> 3) & 31u;
  unsigned int ks   = (idx >> 8) & 15u;
  unsigned int nt   = (idx >> 12) & 31u;
  unsigned int j    = idx >> 17;
  if (j >= J_TOT) return;
  unsigned int n = nt * 16u + (lane & 15u);
  unsigned int k = ks * 32u + ((lane >> 4) << 4) + (v << 1);
  const float* s = W1 + ((size_t)j * D_DIM + k) * D_DIM + n;
  ws1[idx] = pack_bf(s[0], s[D_DIM]);
}

// W2a[13,512,6] / W2b[30,512,3] -> ws2, N zero-padded to 16.
// dword index = ((j*16 + kstep)*256 + lane*8 + v)
__global__ void swizzle_w2(const float* __restrict__ W2a, const float* __restrict__ W2b,
                           unsigned int* __restrict__ ws2) {
  unsigned int idx  = blockIdx.x * 256u + threadIdx.x;     // 43*16*256 total
  unsigned int v    = idx & 7u;
  unsigned int lane = (idx >> 3) & 31u;
  unsigned int ks   = (idx >> 8) & 15u;
  unsigned int j    = idx >> 12;
  if (j >= J_TOT) return;
  unsigned int n = lane & 15u;
  unsigned int k = ks * 32u + ((lane >> 4) << 4) + (v << 1);
  float lo = 0.f, hi = 0.f;
  if (j < J6_CNT) {
    if (n < 6u) { const float* s = W2a + ((size_t)j * D_DIM + k) * 6 + n; lo = s[0]; hi = s[6]; }
  } else {
    if (n < 3u) { const float* s = W2b + ((size_t)(j - J6_CNT) * D_DIM + k) * 3 + n; lo = s[0]; hi = s[3]; }
  }
  ws2[idx] = pack_bf(lo, hi);
}

// --------------------------------------------------------------------------
// Fused decoder. grid = (98 row-tiles, 43 joints), block = 256 (8 waves).
// --------------------------------------------------------------------------
__launch_bounds__(256)
__global__ void motion_decoder(const float* __restrict__ x,
                               const float* __restrict__ b1,
                               const float* __restrict__ b2a,
                               const float* __restrict__ b2b,
                               const unsigned int* __restrict__ ws1,
                               const unsigned int* __restrict__ ws2,
                               float* __restrict__ out) {
  __shared__ __align__(16) unsigned char lds[32 * RSTRIDE * 2 + 8 * 256 * 4];
  unsigned char* ldsA = lds;                                // X tile, bf16
  unsigned char* ldsH = lds + 32 * RSTRIDE;                 // h tile, bf16
  float*         redf = (float*)(lds + 64 * RSTRIDE);       // layer-2 partials

  const int tid  = threadIdx.x;
  const int lane = tid & 31;
  const int w    = tid >> 5;                                // wave id 0..7
  const int j    = blockIdx.y;
  const int row0 = blockIdx.x * MTILE;

  // ---- stage X tile (32 rows x 512) fp32 -> bf16 LDS -------------------
  #pragma unroll
  for (int rep = 0; rep < 16; ++rep) {
    int idx = rep * 256 + tid;                              // 0..4095 float4s
    int row = idx >> 7;
    int c4  = idx & 127;
    f32x4 vv = *(const f32x4*)(x + ((size_t)(row0 + row) * J_TOT + j) * D_DIM + c4 * 4);
    u32x2 p; p.x = pack_bf(vv.x, vv.y); p.y = pack_bf(vv.z, vv.w);
    *(u32x2*)(ldsA + row * RSTRIDE + c4 * 8) = p;
  }
  __syncthreads();

  // ---- layer 1: 32x512x512 GEMM, wave w owns N range [w*64, w*64+64) ---
  v8f acc[2][4];
  #pragma unroll
  for (int mi = 0; mi < 2; ++mi)
    #pragma unroll
    for (int ni = 0; ni < 4; ++ni) acc[mi][ni] = (v8f){0.f,0.f,0.f,0.f,0.f,0.f,0.f,0.f};

  const int arow       = lane & 15;
  const int aChunkOff  = (lane >> 4) << 5;                  // 0 or 32 bytes
  const size_t bBase   = ((size_t)(j * 32 + w * 4) * 16) * 256 + lane * 8;

  for (int ks = 0; ks < 16; ++ks) {
    Frag a[2];
    #pragma unroll
    for (int mi = 0; mi < 2; ++mi) {
      const unsigned char* ap = ldsA + (mi * 16 + arow) * RSTRIDE + ks * 64 + aChunkOff;
      a[mi].q[0] = *(const u32x4*)(ap);
      a[mi].q[1] = *(const u32x4*)(ap + 16);
    }
    #pragma unroll
    for (int ni = 0; ni < 4; ++ni) {
      Frag b;
      const unsigned int* bp = ws1 + bBase + (size_t)ni * (16 * 256) + ks * 256;
      b.q[0] = *(const u32x4*)(bp);
      b.q[1] = *(const u32x4*)(bp + 4);
      acc[0][ni] = __builtin_amdgcn_wmma_f32_16x16x32_bf16(
          false, a[0].v, false, b.v, (short)0, acc[0][ni], false, false);
      acc[1][ni] = __builtin_amdgcn_wmma_f32_16x16x32_bf16(
          false, a[1].v, false, b.v, (short)0, acc[1][ni], false, false);
    }
  }

  // ---- bias + exact GELU, h -> LDS (bf16) ------------------------------
  // C/D layout: col n = lane&15 (same for all 8 VGPRs), row = v + 8*(lane>=16).
  #pragma unroll
  for (int ni = 0; ni < 4; ++ni) {
    int n = w * 64 + ni * 16 + (lane & 15);
    float bias = b1[(size_t)j * D_DIM + n];
    #pragma unroll
    for (int mi = 0; mi < 2; ++mi) {
      #pragma unroll
      for (int v = 0; v < 8; ++v) {
        float h = acc[mi][ni][v] + bias;
        h = 0.5f * h * (1.0f + erff(h * 0.70710678118654752f));
        int r = mi * 16 + v + ((lane >> 4) << 3);
        *(unsigned short*)(ldsH + r * RSTRIDE + n * 2) = (unsigned short)f2bf(h);
      }
    }
  }
  __syncthreads();

  // ---- layer 2: 32x16x512 (N padded to 16); wave = (m, k-quarter) ------
  {
    const int m  = w & 1;
    const int kq = w >> 1;
    v8f acc2 = (v8f){0.f,0.f,0.f,0.f,0.f,0.f,0.f,0.f};
    #pragma unroll
    for (int t4 = 0; t4 < 4; ++t4) {
      int ks2 = kq * 4 + t4;
      Frag a, b;
      const unsigned char* ap = ldsH + (m * 16 + arow) * RSTRIDE + ks2 * 64 + aChunkOff;
      a.q[0] = *(const u32x4*)(ap);
      a.q[1] = *(const u32x4*)(ap + 16);
      const unsigned int* bp = ws2 + ((size_t)(j * 16 + ks2) * 256) + lane * 8;
      b.q[0] = *(const u32x4*)(bp);
      b.q[1] = *(const u32x4*)(bp + 4);
      acc2 = __builtin_amdgcn_wmma_f32_16x16x32_bf16(
          false, a.v, false, b.v, (short)0, acc2, false, false);
    }
    #pragma unroll
    for (int v = 0; v < 8; ++v) redf[w * 256 + v * 32 + lane] = acc2[v];
  }
  __syncthreads();

  // ---- reduce 4 k-partials, add bias, scatter to concat layout ---------
  {
    const int v   = tid >> 5;                               // C-layout VGPR index
    const int l   = tid & 31;
    const int col = l & 15;
    const int rl  = v + ((l >> 4) << 3);
    const int outdim = (j < J6_CNT) ? 6 : 3;
    const int obase  = (j < J6_CNT) ? j * 6 : (J6_CNT * 6 + (j - J6_CNT) * 3);
    if (col < outdim) {
      float bias = (j < J6_CNT) ? b2a[j * 6 + col] : b2b[(j - J6_CNT) * 3 + col];
      #pragma unroll
      for (int m = 0; m < 2; ++m) {
        float s = 0.f;
        #pragma unroll
        for (int kq = 0; kq < 4; ++kq) s += redf[(kq * 2 + m) * 256 + v * 32 + l];
        int r = row0 + m * 16 + rl;
        out[(size_t)r * OUT_STR + obase + col] = s + bias;
      }
    }
  }
}

// --------------------------------------------------------------------------
extern "C" void kernel_launch(void* const* d_in, const int* in_sizes, int n_in,
                              void* d_out, int out_size, void* d_ws, size_t ws_size,
                              hipStream_t stream) {
  const float* x   = (const float*)d_in[0];
  const float* W1  = (const float*)d_in[1];
  const float* b1  = (const float*)d_in[2];
  const float* W2a = (const float*)d_in[3];
  const float* b2a = (const float*)d_in[4];
  const float* W2b = (const float*)d_in[5];
  const float* b2b = (const float*)d_in[6];

  unsigned int* ws1 = (unsigned int*)d_ws;                  // 43*32*16*256 dwords (~21.5 MB)
  unsigned int* ws2 = ws1 + (size_t)J_TOT * 32 * 16 * 256;  // + 43*16*256 dwords (~0.7 MB)

  hipLaunchKernelGGL(swizzle_w1, dim3((J_TOT * 32 * 16 * 256) / 256), dim3(256), 0, stream,
                     W1, ws1);
  hipLaunchKernelGGL(swizzle_w2, dim3((J_TOT * 16 * 256) / 256), dim3(256), 0, stream,
                     W2a, W2b, ws2);
  hipLaunchKernelGGL(motion_decoder, dim3(NROWT, J_TOT), dim3(256), 0, stream,
                     x, b1, b2a, b2b, ws1, ws2, (float*)d_out);
}
